// TypeScorer_61607010894566
// MI455X (gfx1250) — matrix operand order repaired and checked by reading, not compiled, not run on hardware
//
#include <hip/hip_runtime.h>
#include <math.h>

typedef __attribute__((ext_vector_type(4)))  __bf16 v4bf;
typedef __attribute__((ext_vector_type(8)))  __bf16 v8bf;
typedef __attribute__((ext_vector_type(16))) __bf16 v16bf;
typedef __attribute__((ext_vector_type(8)))  float  v8f;

#define BB   256
#define LL   512
#define DD   768
#define HH   1000
#define HPAD 1024      // 64 tiles of 16 (even pair count for 2x2 blocking)
#define K1   2324
#define K1P  2336      // 73 * 32
#define N1   1024
#define N2   512
#define N3   256
#define N4   30
#define N4P  32
#define SLOPE 0.01f

__constant__ int c_bins[15] = {1,2,3,4,5,6,7,8,12,16,20,24,32,64,128};

static __device__ __forceinline__ v16bf cat16(v8bf a, v8bf b) {
  return __builtin_shufflevector(a, b, 0,1,2,3,4,5,6,7,8,9,10,11,12,13,14,15);
}

// ---------- f32 -> bf16 bulk convert (embeds) ----------
__global__ __launch_bounds__(256) void cvt_f32_bf16(const float* __restrict__ in,
                                                    __bf16* __restrict__ out, long n4) {
  long i = ((long)blockIdx.x * 256 + threadIdx.x);
  if (i >= n4) return;
  const float4 f = ((const float4*)in)[i];
  v4bf o = { (__bf16)f.x, (__bf16)f.y, (__bf16)f.z, (__bf16)f.w };
  ((v4bf*)out)[i] = o;
}

// ---------- W[k][n] f32 -> Wt[n][k] bf16 with zero padding ----------
__global__ __launch_bounds__(256) void cvt_transpose(const float* __restrict__ W,
                                                     __bf16* __restrict__ Wt,
                                                     int Kreal, int Kp, int Nreal, int Npad) {
  long id = (long)blockIdx.x * 256 + threadIdx.x;
  long total = (long)Npad * Kp;
  if (id >= total) return;
  int n = (int)(id / Kp);
  int k = (int)(id % Kp);
  float v = (n < Nreal && k < Kreal) ? W[(long)k * Nreal + n] : 0.f;
  Wt[id] = (__bf16)v;
}

// ---------- fused attention-score GEMM, 2x2 register blocking ----------
// attns[m] = sum_j leaky(embeds[m,:]@aw1[:,j] + ab1[j]) * aw2[j] + ab2
#define AG_MT 64
#define AG_ST 776   // padded bf16 LDS row stride (4-bank shift/row)

__global__ __launch_bounds__(256) void attn_gemm(const __bf16* __restrict__ ebf,
                                                 const __bf16* __restrict__ awT,
                                                 const float* __restrict__ ab1,
                                                 const float* __restrict__ aw2,
                                                 const float* __restrict__ ab2,
                                                 float* __restrict__ attns) {
  __shared__ __align__(16) __bf16 As[AG_MT * AG_ST];
  __shared__ float attacc[AG_MT];
  const int t    = threadIdx.x;
  const int wave = t >> 5;
  const int lane = t & 31;
  const int lo   = lane & 15;
  const int hi   = lane >> 4;
  const long rowBase = (long)blockIdx.x * AG_MT;

  if (t < AG_MT) attacc[t] = 0.f;
  // stage 64x768 bf16 panel of A into LDS via the async direct-to-LDS path
  // (GLOBAL_LOAD_ASYNC_TO_LDS_B128: VDST = LDS byte address, VADDR = 64-bit global)
  for (int c = t; c < AG_MT * 96; c += 256) {
    int r = c / 96, k = (c % 96) * 8;
    unsigned ldsAddr = (unsigned)(unsigned long long)(const void*)(&As[r * AG_ST + k]);
    unsigned long long gAddr =
        (unsigned long long)(const void*)(ebf + (rowBase + r) * DD + k);
    asm volatile("global_load_async_to_lds_b128 %0, %1, off"
                 :: "v"(ldsAddr), "v"(gAddr) : "memory");
  }
  asm volatile("s_wait_asynccnt 0x0" ::: "memory");
  __syncthreads();

  // wave -> (pair of M-tiles, quarter of N-tile pairs)
  const int mtBase = (wave & 1) * 2;          // M tiles {0,1} or {2,3}
  const int grp0   = wave >> 1;               // N-pair phase 0..3
  float rowAcc[2][8];
#pragma unroll
  for (int mi = 0; mi < 2; ++mi)
#pragma unroll
    for (int r = 0; r < 8; ++r) rowAcc[mi][r] = 0.f;

  const __bf16* aBase0 = &As[(mtBase * 16 + lo) * AG_ST + hi * 16];
  const __bf16* aBase1 = aBase0 + 16 * AG_ST;

  for (int grp = grp0; grp < (HPAD / 32); grp += 4) {
    const int nt0 = grp * 2, nt1 = nt0 + 1;
    v8f acc00 = {}, acc01 = {}, acc10 = {}, acc11 = {};
    const __bf16* aptr0 = aBase0;
    const __bf16* aptr1 = aBase1;
    const __bf16* bptr0 = awT + (long)(nt0 * 16 + lo) * DD + hi * 16;
    const __bf16* bptr1 = awT + (long)(nt1 * 16 + lo) * DD + hi * 16;
#pragma unroll 2
    for (int ks = 0; ks < DD / 32; ++ks) {
      v16bf av0 = cat16(*(const v8bf*)(aptr0), *(const v8bf*)(aptr0 + 8));
      v16bf av1 = cat16(*(const v8bf*)(aptr1), *(const v8bf*)(aptr1 + 8));
      v16bf bv0 = *(const v16bf*)(bptr0);
      v16bf bv1 = *(const v16bf*)(bptr1);
      __builtin_prefetch(bptr0 + 64, 0, 1);
      __builtin_prefetch(bptr1 + 64, 0, 1);
      acc00 = __builtin_amdgcn_wmma_f32_16x16x32_bf16(false, av0, false, bv0,
                                                      (short)0, acc00, false, false);
      acc01 = __builtin_amdgcn_wmma_f32_16x16x32_bf16(false, av0, false, bv1,
                                                      (short)0, acc01, false, false);
      acc10 = __builtin_amdgcn_wmma_f32_16x16x32_bf16(false, av1, false, bv0,
                                                      (short)0, acc10, false, false);
      acc11 = __builtin_amdgcn_wmma_f32_16x16x32_bf16(false, av1, false, bv1,
                                                      (short)0, acc11, false, false);
      aptr0 += 32; aptr1 += 32; bptr0 += 32; bptr1 += 32;
    }
    // epilogue: bias + leaky + *aw2, reduce across the 16 columns of each tile
#pragma unroll
    for (int nj = 0; nj < 2; ++nj) {
      const int j = (nj ? nt1 : nt0) * 16 + lo;
      float bias = 0.f, a2 = 0.f;
      if (j < HH) { bias = ab1[j]; a2 = aw2[j]; }
#pragma unroll
      for (int mi = 0; mi < 2; ++mi) {
        const v8f& acc = mi ? (nj ? acc11 : acc10) : (nj ? acc01 : acc00);
#pragma unroll
        for (int r = 0; r < 8; ++r) {
          float v = acc[r] + bias;
          v = v > 0.f ? v : SLOPE * v;
          v = (j < HH) ? v * a2 : 0.f;
          v += __shfl_xor(v, 1);
          v += __shfl_xor(v, 2);
          v += __shfl_xor(v, 4);
          v += __shfl_xor(v, 8);   // all lanes in each 16-half hold the row sum
          rowAcc[mi][r] += v;
        }
      }
    }
  }
  if (lo == 0) {
#pragma unroll
    for (int mi = 0; mi < 2; ++mi)
#pragma unroll
      for (int r = 0; r < 8; ++r)
        atomicAdd(&attacc[(mtBase + mi) * 16 + r + hi * 8], rowAcc[mi][r]);
  }
  __syncthreads();
  if (t < AG_MT) attns[rowBase + t] = attacc[t] + ab2[0];
}

// ---------- masked softmax + weighted embed reduction ----------
__global__ __launch_bounds__(256) void softmax_attn(const float* __restrict__ attns,
                                                    const long long* __restrict__ spans,
                                                    const float* __restrict__ embeds,
                                                    float* __restrict__ attne) {
  __shared__ float wsm[LL];
  __shared__ float red[256];
  const int b = blockIdx.x, t = threadIdx.x;
  const int s0 = (int)spans[2 * b], e0 = (int)spans[2 * b + 1];
  const float a0 = attns[b * LL + t];
  const float a1 = attns[b * LL + t + 256];
  const bool m0 = (t >= s0) && (t <= e0);
  const bool m1 = (t + 256 >= s0) && (t + 256 <= e0);
  red[t] = fmaxf(m0 ? a0 : -3.0e38f, m1 ? a1 : -3.0e38f);
  __syncthreads();
  for (int s = 128; s > 0; s >>= 1) { if (t < s) red[t] = fmaxf(red[t], red[t + s]); __syncthreads(); }
  const float mx = red[0];
  __syncthreads();
  const float e0v = m0 ? expf(a0 - mx) : 0.f;
  const float e1v = m1 ? expf(a1 - mx) : 0.f;
  red[t] = e0v + e1v;
  __syncthreads();
  for (int s = 128; s > 0; s >>= 1) { if (t < s) red[t] += red[t + s]; __syncthreads(); }
  const float inv = 1.f / red[0];
  wsm[t] = e0v * inv;
  wsm[t + 256] = e1v * inv;
  __syncthreads();
  const float* eb = embeds + (long)b * LL * DD;
  for (int d = t; d < DD; d += 256) {
    float acc = 0.f;
    for (int l = s0; l <= e0; ++l) acc += wsm[l] * eb[(long)l * DD + d];
    attne[(long)b * DD + d] = acc;
  }
}

// ---------- build concat feature (bf16, K-padded) ----------
__global__ __launch_bounds__(256) void build_x(const float* __restrict__ embeds,
                                               const long long* __restrict__ spans,
                                               const float* __restrict__ attne,
                                               const float* __restrict__ wemb,
                                               __bf16* __restrict__ xb) {
  const int b = blockIdx.x, t = threadIdx.x;
  const int s0 = (int)spans[2 * b], e0 = (int)spans[2 * b + 1];
  const int width = e0 - s0 + 1;
  int idx = 0;
#pragma unroll
  for (int i = 0; i < 15; ++i) idx += (width > c_bins[i]) ? 1 : 0;
  const float* eb = embeds + (long)b * LL * DD;
  for (int c = t; c < K1P; c += 256) {
    float v;
    if (c < DD)            v = eb[(long)s0 * DD + c];
    else if (c < 2 * DD)   v = eb[(long)e0 * DD + (c - DD)];
    else if (c < 3 * DD)   v = attne[(long)b * DD + (c - 2 * DD)];
    else if (c < K1)       v = wemb[idx * 20 + (c - 3 * DD)];
    else                   v = 0.f;
    xb[(long)b * K1P + c] = (__bf16)v;
  }
}

// ---------- generic WMMA MLP layer: Y = act(X @ Wt^T + b) ----------
__global__ __launch_bounds__(256) void mlp_gemm(const __bf16* __restrict__ X,
                                                const __bf16* __restrict__ Wt,
                                                const float* __restrict__ bias,
                                                int M, int Kp, int Npad, int Nreal, int act,
                                                __bf16* __restrict__ outBf, int obStride,
                                                float* __restrict__ outF, int ofStride) {
  const int wave = threadIdx.x >> 5;
  const int lane = threadIdx.x & 31;
  const int lo = lane & 15, hi = lane >> 4;
  const int ntiles = Npad >> 4;
  const int tile = blockIdx.x * 8 + wave;
  const int mt = tile / ntiles, nt = tile % ntiles;
  if (mt * 16 >= M) return;                         // wave-uniform
  const __bf16* aptr = X + (long)(mt * 16 + lo) * Kp + hi * 16;
  const __bf16* bptr = Wt + (long)(nt * 16 + lo) * Kp + hi * 16;
  v8f acc = {};
  for (int ks = 0; ks < (Kp >> 5); ++ks) {
    v16bf av = cat16(*(const v8bf*)(aptr), *(const v8bf*)(aptr + 8));
    v16bf bv = *(const v16bf*)(bptr);
    acc = __builtin_amdgcn_wmma_f32_16x16x32_bf16(false, av, false, bv,
                                                  (short)0, acc, false, false);
    aptr += 32; bptr += 32;
  }
  const int j = nt * 16 + lo;
  if (j < Nreal) {
    const float bv = bias[j];
#pragma unroll
    for (int r = 0; r < 8; ++r) {
      const int row = mt * 16 + r + hi * 8;
      float v = acc[r] + bv;
      if (act) v = v > 0.f ? v : SLOPE * v;
      if (outBf) outBf[(long)row * obStride + j] = (__bf16)v;
      if (outF)  outF[(long)row * ofStride + j] = v;
    }
  }
}

extern "C" void kernel_launch(void* const* d_in, const int* in_sizes, int n_in,
                              void* d_out, int out_size, void* d_ws, size_t ws_size,
                              hipStream_t stream) {
  const float*      embeds = (const float*)d_in[0];
  const long long*  spans  = (const long long*)d_in[1];
  const float*      aw1    = (const float*)d_in[2];
  const float*      ab1    = (const float*)d_in[3];
  const float*      aw2    = (const float*)d_in[4];
  const float*      ab2    = (const float*)d_in[5];
  const float*      wemb   = (const float*)d_in[6];
  const float*      sw1    = (const float*)d_in[7];
  const float*      sb1    = (const float*)d_in[8];
  const float*      sw2    = (const float*)d_in[9];
  const float*      sb2    = (const float*)d_in[10];
  const float*      sw3    = (const float*)d_in[11];
  const float*      sb3    = (const float*)d_in[12];
  const float*      sw4    = (const float*)d_in[13];
  const float*      sb4    = (const float*)d_in[14];

  char* ws = (char*)d_ws;
  size_t off = 0;
  auto take = [&](size_t bytes) -> char* {
    char* p = ws + off;
    off = (off + bytes + 255) & ~(size_t)255;
    return p;
  };
  const long ELD = (long)BB * LL * DD;
  __bf16* ebf   = (__bf16*)take(ELD * 2);
  __bf16* awT   = (__bf16*)take((size_t)HPAD * DD * 2);
  float*  attns = (float*) take((size_t)BB * LL * 4);
  float*  attne = (float*) take((size_t)BB * DD * 4);
  __bf16* xb    = (__bf16*)take((size_t)BB * K1P * 2);
  __bf16* w1T   = (__bf16*)take((size_t)N1 * K1P * 2);
  __bf16* w2T   = (__bf16*)take((size_t)N2 * N1 * 2);
  __bf16* w3T   = (__bf16*)take((size_t)N3 * N2 * 2);
  __bf16* w4T   = (__bf16*)take((size_t)N4P * N3 * 2);
  __bf16* a1b   = (__bf16*)take((size_t)BB * N1 * 2);
  __bf16* a2b   = (__bf16*)take((size_t)BB * N2 * 2);
  __bf16* a3b   = (__bf16*)take((size_t)BB * N3 * 2);

  // 1) bf16 conversions
  {
    long n4 = ELD / 4;
    cvt_f32_bf16<<<(int)((n4 + 255) / 256), 256, 0, stream>>>(embeds, ebf, n4);
  }
  auto launch_cvt = [&](const float* W, __bf16* Wt, int Kr, int Kp, int Nr, int Np) {
    long tot = (long)Np * Kp;
    cvt_transpose<<<(int)((tot + 255) / 256), 256, 0, stream>>>(W, Wt, Kr, Kp, Nr, Np);
  };
  launch_cvt(aw1, awT, DD, DD, HH, HPAD);
  launch_cvt(sw1, w1T, K1, K1P, N1, N1);
  launch_cvt(sw2, w2T, N1, N1, N2, N2);
  launch_cvt(sw3, w3T, N2, N2, N3, N3);
  launch_cvt(sw4, w4T, N3, N3, N4, N4P);

  // 2) fused attention-score GEMM (dominant WMMA kernel)
  attn_gemm<<<(BB * LL) / AG_MT, 256, 0, stream>>>(ebf, awT, ab1, aw2, ab2, attns);

  // 3) masked softmax + weighted reduce
  softmax_attn<<<BB, 256, 0, stream>>>(attns, spans, embeds, attne);

  // 4) concat feature
  build_x<<<BB, 256, 0, stream>>>(embeds, spans, attne, wemb, xb);

  // 5) MLP (WMMA)
  auto launch_mlp = [&](const __bf16* X, const __bf16* Wt, const float* b,
                        int Kp, int Np, int Nr, int act,
                        __bf16* ob, int obs, float* of, int ofs) {
    int tiles = (BB / 16) * (Np / 16);
    mlp_gemm<<<(tiles + 7) / 8, 256, 0, stream>>>(X, Wt, b, BB, Kp, Np, Nr, act,
                                                  ob, obs, of, ofs);
  };
  launch_mlp(xb,  w1T, sb1, K1P, N1,  N1, 1, a1b, N1, nullptr, 0);
  launch_mlp(a1b, w2T, sb2, N1,  N2,  N2, 1, a2b, N2, nullptr, 0);
  launch_mlp(a2b, w3T, sb3, N2,  N3,  N3, 1, a3b, N3, nullptr, 0);
  launch_mlp(a3b, w4T, sb4, N3,  N4P, N4, 0, nullptr, 0, (float*)d_out, N4);
}
